// Block_58772332478734
// MI455X (gfx1250) — matrix operand order repaired and checked by reading, not compiled, not run on hardware
//
#include <hip/hip_runtime.h>

// ---------------------------------------------------------------------------
// ViT encoder block for MI455X (gfx1250, wave32, WMMA bf16 16x16x32).
// All GEMMs + attention matmuls run on v_wmma_f32_16x16x32_bf16 with f32 acc.
// GEMMs are register-blocked 4x2 per wave (64x32 C block): 8 WMMAs per
// 6 fragment loads per K=32 step (1 WMMA : 1.5 b128 loads).
// ---------------------------------------------------------------------------

typedef __attribute__((ext_vector_type(16))) __bf16       v16bf;
typedef __attribute__((ext_vector_type(8)))  float        v8f;
typedef __attribute__((ext_vector_type(4)))  unsigned int u32x4;

union Frag { v16bf v; u32x4 q[2]; };

static __device__ __forceinline__ v8f wmma_bf16(const Frag& a, const Frag& b, v8f c) {
    return __builtin_amdgcn_wmma_f32_16x16x32_bf16(false, a.v, false, b.v,
                                                   (short)0, c, false, false);
}

// ---------------------------------------------------------------------------
// Wave-level 64x32 register-blocked GEMM: acc[4][2] 16x16 f32 tiles.
// A row-major [M,K] bf16; B supplied as BT row-major [N,K] bf16.
// Fragment layouts per CDNA5 ISA 7.12.2 (wave32):
//   A: lane row = lane%16, K elems = 8*(lane/16)+{0..7} and 16+8*(lane/16)+{0..7}
//   B: lane col = lane%16, K elems = 8*(lane/16)+{0..7} and 16+8*(lane/16)+{0..7}
//   C: lane col = lane%16, rows = 8*(lane/16)+r, r=0..7
// ---------------------------------------------------------------------------
static __device__ __forceinline__ void gemm_tile_4x2(const __bf16* __restrict__ A,
                                                     const __bf16* __restrict__ BT,
                                                     int row0, int col0, int K,
                                                     v8f acc[4][2]) {
    const int lane = threadIdx.x & 31;
    const int rl = lane & 15, hi = lane >> 4;
    const __bf16* arow[4];
    const __bf16* brow[2];
#pragma unroll
    for (int i = 0; i < 4; ++i) arow[i] = A  + (size_t)(row0 + i * 16 + rl) * K;
#pragma unroll
    for (int j = 0; j < 2; ++j) brow[j] = BT + (size_t)(col0 + j * 16 + rl) * K;
    for (int k0 = 0; k0 < K; k0 += 32) {
        Frag a[4], b[2];
#pragma unroll
        for (int i = 0; i < 4; ++i) {
            a[i].q[0] = *(const u32x4*)(arow[i] + k0 + hi * 8);
            a[i].q[1] = *(const u32x4*)(arow[i] + k0 + hi * 8 + 16);
        }
#pragma unroll
        for (int j = 0; j < 2; ++j) {
            b[j].q[0] = *(const u32x4*)(brow[j] + k0 + hi * 8);
            b[j].q[1] = *(const u32x4*)(brow[j] + k0 + hi * 8 + 16);
        }
#pragma unroll
        for (int i = 0; i < 4; ++i)
#pragma unroll
            for (int j = 0; j < 2; ++j)
                acc[i][j] = wmma_bf16(a[i], b[j], acc[i][j]);
    }
}

// Block = 256 threads = 8 waves arranged 2 (rows) x 4 (cols) of 64x32 blocks
// -> macro tile 128 x 128.
static __device__ __forceinline__ void block_tile(int& row0, int& col0) {
    const int wid = threadIdx.x >> 5;
    row0 = blockIdx.y * 128 + (wid & 1) * 64;
    col0 = blockIdx.x * 128 + (wid >> 1) * 32;
}

// ---------------------------------------------------------------------------
// Transpose-convert f32 W[K,N] -> bf16 WT[N,K]
// ---------------------------------------------------------------------------
__global__ __launch_bounds__(256) void k_tconv(const float* __restrict__ w,
                                               __bf16* __restrict__ wT,
                                               int K, int N) {
    size_t i = (size_t)blockIdx.x * 256 + threadIdx.x;
    size_t total = (size_t)K * N;
    if (i >= total) return;
    size_t n = i / (size_t)K, kk = i % (size_t)K;
    wT[i] = (__bf16)w[kk * (size_t)N + n];
}

// ---------------------------------------------------------------------------
// LayerNorm over rows of 1024, f32 in -> bf16 out. One block (256 thr) / row.
// ---------------------------------------------------------------------------
__global__ __launch_bounds__(256) void k_ln(const float* __restrict__ x,
                                            const float* __restrict__ g,
                                            const float* __restrict__ b,
                                            __bf16* __restrict__ out) {
    __shared__ float sm[16];
    const int row = blockIdx.x, tid = threadIdx.x;
    const float* xr = x + (size_t)row * 1024;
    float v[4], s = 0.f, s2 = 0.f;
#pragma unroll
    for (int i = 0; i < 4; ++i) {
        v[i] = xr[tid + i * 256];
        s += v[i]; s2 += v[i] * v[i];
    }
#pragma unroll
    for (int mk = 16; mk >= 1; mk >>= 1) {
        s  += __shfl_xor(s,  mk, 32);
        s2 += __shfl_xor(s2, mk, 32);
    }
    const int wid = tid >> 5, lane = tid & 31;
    if (lane == 0) { sm[wid] = s; sm[wid + 8] = s2; }
    __syncthreads();
    if (tid < 32) {
        float a = (lane < 8) ? sm[lane] : 0.f;
        float c = (lane < 8) ? sm[lane + 8] : 0.f;
#pragma unroll
        for (int mk = 4; mk >= 1; mk >>= 1) {
            a += __shfl_xor(a, mk, 32);
            c += __shfl_xor(c, mk, 32);
        }
        if (lane == 0) { sm[0] = a; sm[1] = c; }
    }
    __syncthreads();
    const float mean = sm[0] * (1.f / 1024.f);
    const float var  = sm[1] * (1.f / 1024.f) - mean * mean;
    const float rstd = rsqrtf(var + 1e-6f);
#pragma unroll
    for (int i = 0; i < 4; ++i) {
        int c = tid + i * 256;
        out[(size_t)row * 1024 + c] = (__bf16)((v[i] - mean) * rstd * g[c] + b[c]);
    }
}

// ---------------------------------------------------------------------------
// QKV GEMM: [4096,1024] x [1024,3072], scatter into q/k (head-major) and
// transposed vT so attention B-fragments are contiguous loads.
// ---------------------------------------------------------------------------
__global__ __launch_bounds__(256) void k_gemm_qkv(const __bf16* __restrict__ A,
                                                  const __bf16* __restrict__ BT,
                                                  __bf16* __restrict__ qm,
                                                  __bf16* __restrict__ km,
                                                  __bf16* __restrict__ vT) {
    int row0, col0;
    block_tile(row0, col0);
    v8f acc[4][2] = {};
    gemm_tile_4x2(A, BT, row0, col0, 1024, acc);
    const int lane = threadIdx.x & 31;
    const int n = lane & 15, hi = lane >> 4;
#pragma unroll
    for (int i = 0; i < 4; ++i)
#pragma unroll
        for (int j = 0; j < 2; ++j)
#pragma unroll
            for (int rr = 0; rr < 8; ++rr) {
                const int row = row0 + i * 16 + rr + 8 * hi;   // token 0..4095
                const int c   = col0 + j * 16 + n;             // 0..3071
                const __bf16 val = (__bf16)acc[i][j][rr];
                const int bb = row >> 10, nq = row & 1023;
                const int which = c >> 10, cc = c & 1023;
                const int h = cc >> 6, d = cc & 63;
                const size_t bh = (size_t)(bb * 16 + h);
                if (which == 0)      qm[bh * 65536 + (size_t)nq * 64 + d] = val;
                else if (which == 1) km[bh * 65536 + (size_t)nq * 64 + d] = val;
                else                 vT[bh * 65536 + (size_t)d * 1024 + nq] = val;
            }
}

// ---------------------------------------------------------------------------
// Attention: one wave per 16 query rows of one (b,h). Two-pass softmax:
// pass1 streaming row max/sum; pass2 recompute S, P via LDS -> P.V with WMMA.
// ---------------------------------------------------------------------------
__global__ __launch_bounds__(256) void k_attn(const __bf16* __restrict__ q,
                                              const __bf16* __restrict__ k,
                                              const __bf16* __restrict__ vT,
                                              __bf16* __restrict__ o_out) {
    const int N = 1024, DH = 64;
    __shared__ __bf16 plds[8 * 16 * 32];          // per-wave 16x32 P tile

    const int wid = threadIdx.x >> 5, lane = threadIdx.x & 31;
    const int bh = blockIdx.x >> 3, qb = blockIdx.x & 7;
    const int qrow0 = qb * 128 + wid * 16;
    const int rl = lane & 15, hi = lane >> 4;
    const __bf16* qp = q  + (size_t)bh * N * DH;
    const __bf16* kp = k  + (size_t)bh * N * DH;
    const __bf16* vp = vT + (size_t)bh * DH * N;
    const float scale = 0.125f;                    // 64^-0.5

    // Q A-fragments (K = head dim 64 -> two 16x32 fragments), kept in regs.
    Frag aq0, aq1;
    const __bf16* qrow = qp + (size_t)(qrow0 + rl) * DH;
    aq0.q[0] = *(const u32x4*)(qrow + hi * 8);
    aq0.q[1] = *(const u32x4*)(qrow + hi * 8 + 16);
    aq1.q[0] = *(const u32x4*)(qrow + 32 + hi * 8);
    aq1.q[1] = *(const u32x4*)(qrow + 32 + hi * 8 + 16);

    float m[8], l[8];
#pragma unroll
    for (int r = 0; r < 8; ++r) { m[r] = -1e30f; l[r] = 0.f; }

    // -------- pass 1: row max & sum of exp (16 keys / iteration) --------
    for (int kb = 0; kb < 64; ++kb) {
        Frag bk0, bk1;
        const __bf16* krow = kp + (size_t)(kb * 16 + rl) * DH;
        bk0.q[0] = *(const u32x4*)(krow + hi * 8);
        bk0.q[1] = *(const u32x4*)(krow + hi * 8 + 16);
        bk1.q[0] = *(const u32x4*)(krow + 32 + hi * 8);
        bk1.q[1] = *(const u32x4*)(krow + 32 + hi * 8 + 16);
        v8f acc = {};
        acc = wmma_bf16(aq0, bk0, acc);
        acc = wmma_bf16(aq1, bk1, acc);
#pragma unroll
        for (int r = 0; r < 8; ++r) {
            float s = acc[r] * scale;
            float bm = s;
#pragma unroll
            for (int mk = 1; mk < 16; mk <<= 1)
                bm = fmaxf(bm, __shfl_xor(bm, mk, 32));
            const float mn = fmaxf(m[r], bm);
            float e = __expf(s - mn);
#pragma unroll
            for (int mk = 1; mk < 16; mk <<= 1)
                e += __shfl_xor(e, mk, 32);
            l[r] = l[r] * __expf(m[r] - mn) + e;
            m[r] = mn;
        }
    }

    // -------- pass 2: O = softmax(S) . V (32 keys / iteration) --------
    v8f ov[4] = {};
    __bf16* myp = &plds[wid * 512];
    for (int kb = 0; kb < 32; ++kb) {
        v8f s0 = {}, s1 = {};
        {
            Frag bk0, bk1;
            const __bf16* krow = kp + (size_t)(kb * 32 + rl) * DH;
            bk0.q[0] = *(const u32x4*)(krow + hi * 8);
            bk0.q[1] = *(const u32x4*)(krow + hi * 8 + 16);
            bk1.q[0] = *(const u32x4*)(krow + 32 + hi * 8);
            bk1.q[1] = *(const u32x4*)(krow + 32 + hi * 8 + 16);
            s0 = wmma_bf16(aq0, bk0, s0);
            s0 = wmma_bf16(aq1, bk1, s0);
        }
        {
            Frag bk0, bk1;
            const __bf16* krow = kp + (size_t)(kb * 32 + 16 + rl) * DH;
            bk0.q[0] = *(const u32x4*)(krow + hi * 8);
            bk0.q[1] = *(const u32x4*)(krow + hi * 8 + 16);
            bk1.q[0] = *(const u32x4*)(krow + 32 + hi * 8);
            bk1.q[1] = *(const u32x4*)(krow + 32 + hi * 8 + 16);
            s1 = wmma_bf16(aq0, bk0, s1);
            s1 = wmma_bf16(aq1, bk1, s1);
        }
        // exponentiate, stash 16x32 P tile (row-major) in wave-private LDS
#pragma unroll
        for (int r = 0; r < 8; ++r) {
            const float p0 = __expf(s0[r] * scale - m[r]);
            const float p1 = __expf(s1[r] * scale - m[r]);
            myp[(r + 8 * hi) * 32 + rl]      = (__bf16)p0;
            myp[(r + 8 * hi) * 32 + 16 + rl] = (__bf16)p1;
        }
        // reload as A-fragment (same-wave LDS ops are in order; compiler waits)
        Frag ap;
        ap.q[0] = *(const u32x4*)(myp + rl * 32 + hi * 8);
        ap.q[1] = *(const u32x4*)(myp + rl * 32 + hi * 8 + 16);
        // P[16x32] . V[32x64] via vT rows (contiguous over keys)
#pragma unroll
        for (int t = 0; t < 4; ++t) {
            Frag bv;
            const __bf16* vrow = vp + (size_t)(t * 16 + rl) * N + kb * 32;
            bv.q[0] = *(const u32x4*)(vrow + hi * 8);
            bv.q[1] = *(const u32x4*)(vrow + hi * 8 + 16);
            ov[t] = wmma_bf16(ap, bv, ov[t]);
        }
    }

    // write O / l as bf16 into [token, h*64+d]
    const int bb = bh >> 4, h = bh & 15;
#pragma unroll
    for (int t = 0; t < 4; ++t) {
#pragma unroll
        for (int r = 0; r < 8; ++r) {
            const int qr = qrow0 + r + 8 * hi;
            const int ch = h * 64 + t * 16 + rl;
            o_out[((size_t)(bb * 1024 + qr)) * 1024 + ch] = (__bf16)(ov[t][r] / l[r]);
        }
    }
}

// ---------------------------------------------------------------------------
// Proj GEMM + bias + residual -> f32 x1
// ---------------------------------------------------------------------------
__global__ __launch_bounds__(256) void k_gemm_proj(const __bf16* __restrict__ A,
                                                   const __bf16* __restrict__ BT,
                                                   const float* __restrict__ bias,
                                                   const float* __restrict__ xres,
                                                   float* __restrict__ x1) {
    int row0, col0;
    block_tile(row0, col0);
    v8f acc[4][2] = {};
    gemm_tile_4x2(A, BT, row0, col0, 1024, acc);
    const int lane = threadIdx.x & 31;
    const int n = lane & 15, hi = lane >> 4;
#pragma unroll
    for (int i = 0; i < 4; ++i)
#pragma unroll
        for (int j = 0; j < 2; ++j)
#pragma unroll
            for (int rr = 0; rr < 8; ++rr) {
                const int row = row0 + i * 16 + rr + 8 * hi;
                const int c = col0 + j * 16 + n;
                x1[(size_t)row * 1024 + c] =
                    xres[(size_t)row * 1024 + c] + acc[i][j][rr] + bias[c];
            }
}

// ---------------------------------------------------------------------------
// FC1 GEMM + bias + exact-erf GELU -> bf16
// ---------------------------------------------------------------------------
__global__ __launch_bounds__(256) void k_gemm_fc1(const __bf16* __restrict__ A,
                                                  const __bf16* __restrict__ BT,
                                                  const float* __restrict__ bias,
                                                  __bf16* __restrict__ hm) {
    int row0, col0;
    block_tile(row0, col0);
    v8f acc[4][2] = {};
    gemm_tile_4x2(A, BT, row0, col0, 1024, acc);
    const int lane = threadIdx.x & 31;
    const int n = lane & 15, hi = lane >> 4;
#pragma unroll
    for (int i = 0; i < 4; ++i)
#pragma unroll
        for (int j = 0; j < 2; ++j)
#pragma unroll
            for (int rr = 0; rr < 8; ++rr) {
                const int row = row0 + i * 16 + rr + 8 * hi;
                const int c = col0 + j * 16 + n;
                const float t = acc[i][j][rr] + bias[c];
                const float g = 0.5f * t * (1.0f + erff(t * 0.70710678118654752f));
                hm[(size_t)row * 4096 + c] = (__bf16)g;
            }
}

// ---------------------------------------------------------------------------
// FC2 GEMM + bias + residual -> f32 out
// ---------------------------------------------------------------------------
__global__ __launch_bounds__(256) void k_gemm_fc2(const __bf16* __restrict__ A,
                                                  const __bf16* __restrict__ BT,
                                                  const float* __restrict__ bias,
                                                  const float* __restrict__ x1,
                                                  float* __restrict__ out) {
    int row0, col0;
    block_tile(row0, col0);
    v8f acc[4][2] = {};
    gemm_tile_4x2(A, BT, row0, col0, 4096, acc);
    const int lane = threadIdx.x & 31;
    const int n = lane & 15, hi = lane >> 4;
#pragma unroll
    for (int i = 0; i < 4; ++i)
#pragma unroll
        for (int j = 0; j < 2; ++j)
#pragma unroll
            for (int rr = 0; rr < 8; ++rr) {
                const int row = row0 + i * 16 + rr + 8 * hi;
                const int c = col0 + j * 16 + n;
                out[(size_t)row * 1024 + c] =
                    x1[(size_t)row * 1024 + c] + acc[i][j][rr] + bias[c];
            }
}

// ---------------------------------------------------------------------------
extern "C" void kernel_launch(void* const* d_in, const int* in_sizes, int n_in,
                              void* d_out, int out_size, void* d_ws, size_t ws_size,
                              hipStream_t stream) {
    (void)in_sizes; (void)n_in; (void)out_size; (void)ws_size;
    const float* x      = (const float*)d_in[0];
    const float* ln1_g  = (const float*)d_in[1];
    const float* ln1_b  = (const float*)d_in[2];
    const float* ln2_g  = (const float*)d_in[3];
    const float* ln2_b  = (const float*)d_in[4];
    const float* w_qkv  = (const float*)d_in[5];
    const float* w_proj = (const float*)d_in[6];
    const float* b_proj = (const float*)d_in[7];
    const float* w_fc1  = (const float*)d_in[8];
    const float* b_fc1  = (const float*)d_in[9];
    const float* w_fc2  = (const float*)d_in[10];
    const float* b_fc2  = (const float*)d_in[11];
    float* out = (float*)d_out;

    // workspace layout (~84 MB)
    char* ws = (char*)d_ws;
    __bf16* wqkvT = (__bf16*)ws;  ws += (size_t)3072 * 1024 * 2;
    __bf16* wprojT = (__bf16*)ws; ws += (size_t)1024 * 1024 * 2;
    __bf16* wfc1T = (__bf16*)ws;  ws += (size_t)4096 * 1024 * 2;
    __bf16* wfc2T = (__bf16*)ws;  ws += (size_t)1024 * 4096 * 2;
    __bf16* h = (__bf16*)ws;      ws += (size_t)4096 * 1024 * 2;   // ln1 out, later ln2 out
    char* big = ws;               ws += (size_t)4096 * 4096 * 2;   // q|k|vT|attn_o, later hm
    __bf16* qm = (__bf16*)big;
    __bf16* km = (__bf16*)(big + (size_t)1 * 4096 * 1024 * 2);
    __bf16* vT = (__bf16*)(big + (size_t)2 * 4096 * 1024 * 2);
    __bf16* ao = (__bf16*)(big + (size_t)3 * 4096 * 1024 * 2);
    __bf16* hm = (__bf16*)big;                                     // reuse after attention
    float* x1 = (float*)ws;       ws += (size_t)4096 * 1024 * 4;

    // weight transpose-converts (f32 [K,N] -> bf16 [N,K])
    k_tconv<<<(3072 * 1024 + 255) / 256, 256, 0, stream>>>(w_qkv,  wqkvT, 1024, 3072);
    k_tconv<<<(1024 * 1024 + 255) / 256, 256, 0, stream>>>(w_proj, wprojT, 1024, 1024);
    k_tconv<<<(4096 * 1024 + 255) / 256, 256, 0, stream>>>(w_fc1,  wfc1T, 1024, 4096);
    k_tconv<<<(4096 * 1024 + 255) / 256, 256, 0, stream>>>(w_fc2,  wfc2T, 4096, 1024);

    // attention sub-block
    k_ln<<<4096, 256, 0, stream>>>(x, ln1_g, ln1_b, h);
    k_gemm_qkv<<<dim3(24, 32), 256, 0, stream>>>(h, wqkvT, qm, km, vT);
    k_attn<<<512, 256, 0, stream>>>(qm, km, vT, ao);
    k_gemm_proj<<<dim3(8, 32), 256, 0, stream>>>(ao, wprojT, b_proj, x, x1);

    // MLP sub-block
    k_ln<<<4096, 256, 0, stream>>>(x1, ln2_g, ln2_b, h);
    k_gemm_fc1<<<dim3(32, 32), 256, 0, stream>>>(h, wfc1T, b_fc1, hm);
    k_gemm_fc2<<<dim3(8, 32), 256, 0, stream>>>(hm, wfc2T, b_fc2, x1, out);
}
